// GlobalBlock_120259084730
// MI455X (gfx1250) — matrix operand order repaired and checked by reading, not compiled, not run on hardware
//
#include <hip/hip_runtime.h>

// ---------------------------------------------------------------------------
// MLP-Mixer-ish block, fully fused for MI455X (gfx1250, wave32, WMMA).
// T=6, BS=16384, C=192, H=4, DH=48.  One workgroup = 16 batch elems = 96 rows.
// Big GEMMs -> v_wmma_f32_16x16x32_bf16 (bf16 in, f32 accum), M-blocked x3 so
// each B fragment (1KB from L2) feeds three WMMAs (3x less L2 traffic).
// Weights pre-swizzled to WMMA B-fragment order in d_ws by pack kernels.
// ---------------------------------------------------------------------------

#define TT   6
#define BSZ  16384
#define CC   192
#define NH   4
#define DHD  48
#define BT   16          // batch elements per workgroup
#define MR   96          // rows per tile = TT*BT
#define SF   197         // f32 LDS stride (odd -> bank spread)
#define SB   200         // bf16 LDS stride (8*odd -> 16B-chunk spread)
#define SH   392         // bf16 stride for FFN hidden (8*odd)

typedef __attribute__((ext_vector_type(16))) __bf16 v16bf;
typedef __attribute__((ext_vector_type(8)))  __bf16 v8bf;
typedef __attribute__((ext_vector_type(8)))  float  v8f;

__device__ __forceinline__ float gelu_f(float v) {
  return 0.5f * v * (1.0f + erff(v * 0.7071067811865476f));
}

// A fragment (16x32 bf16, ISA 7.12.2): lane l -> row m=l&15, half=l>>4.
// element i<8 : k = half*8 + i ; element i>=8 : k = 16 + half*8 + (i-8)
// => two contiguous 16-byte chunks at columns half*8 and half*8+16.
__device__ __forceinline__ v16bf load_a_frag(const __bf16* tile, int stride, int lane) {
  const int m = lane & 15;
  const int h = lane >> 4;
  const __bf16* p = tile + m * stride + h * 8;
  union { v16bf v; v8bf q[2]; } u;
  u.q[0] = *reinterpret_cast<const v8bf*>(p);
  u.q[1] = *reinterpret_cast<const v8bf*>(p + 16);
  return u.v;
}

// M=96 GEMM: 8 waves over (6/3)*ntiles supertiles; each supertile = one N-tile
// x three M-tiles sharing every B fragment (loaded once per k-step).
template <typename EP>
__device__ __forceinline__ void run_gemm(const __bf16* A, int sA, int ktiles,
                                         const __bf16* Wp, int ntiles,
                                         int wave, int lane, EP ep) {
  const int nsuper = 2 * ntiles;
  for (int st = wave; st < nsuper; st += 8) {
    const int g6  = st / ntiles;         // 0 or 1
    const int nt  = st - g6 * ntiles;
    const int mt0 = g6 * 3;
    v8f acc0 = {}, acc1 = {}, acc2 = {};
    const __bf16* bp = Wp + (((size_t)nt) << 9) + (lane << 4);
    for (int kt = 0; kt < ktiles; ++kt) {
      v16bf b = *reinterpret_cast<const v16bf*>(bp + (((size_t)kt * ntiles) << 9));
      const __bf16* abase = A + kt * 32;
      v16bf a0 = load_a_frag(abase + (mt0 * 16) * sA, sA, lane);
      acc0 = __builtin_amdgcn_wmma_f32_16x16x32_bf16(false, a0, false, b, (short)0, acc0, false, false);
      v16bf a1 = load_a_frag(abase + ((mt0 + 1) * 16) * sA, sA, lane);
      acc1 = __builtin_amdgcn_wmma_f32_16x16x32_bf16(false, a1, false, b, (short)0, acc1, false, false);
      v16bf a2 = load_a_frag(abase + ((mt0 + 2) * 16) * sA, sA, lane);
      acc2 = __builtin_amdgcn_wmma_f32_16x16x32_bf16(false, a2, false, b, (short)0, acc2, false, false);
    }
    ep(mt0,     nt, acc0);
    ep(mt0 + 1, nt, acc1);
    ep(mt0 + 2, nt, acc2);
  }
}

// Pack f32 weight W[K][N] into per-fragment bf16 B layout:
// dst[f*512 + l*16 + e] = W[kt*32 + (l>>4)*8 + (e&7) + (e>>3)*16][nt*16 + (l&15)]
__global__ __launch_bounds__(256)
void pack_w(const float* __restrict__ src, __bf16* __restrict__ dst, int K, int N) {
  int idx = blockIdx.x * 256 + threadIdx.x;
  if (idx >= K * N) return;
  int e = idx & 15;
  int l = (idx >> 4) & 31;
  int f = idx >> 9;
  int ntiles = N >> 4;
  int kt = f / ntiles, nt = f - kt * ntiles;
  int k = kt * 32 + (l >> 4) * 8 + (e & 7) + (e >> 3) * 16;
  int n = nt * 16 + (l & 15);
  dst[idx] = (__bf16)src[k * N + n];
}

__global__ __launch_bounds__(256)
void mixer_block(const float* __restrict__ x,
                 const float* __restrict__ tw1, const float* __restrict__ tb1,
                 const float* __restrict__ tw2, const float* __restrict__ tb2,
                 const float* __restrict__ qb,  const float* __restrict__ ab,
                 const float* __restrict__ cb,
                 const float* __restrict__ lng, const float* __restrict__ lnb,
                 const float* __restrict__ f1b, const float* __restrict__ f2b,
                 const float* __restrict__ fng, const float* __restrict__ fnb,
                 const __bf16* __restrict__ qw_p, const __bf16* __restrict__ aw_p,
                 const __bf16* __restrict__ cw_p, const __bf16* __restrict__ f1_p,
                 const __bf16* __restrict__ f2_p,
                 float* __restrict__ out) {
  extern __shared__ char smem_raw[];
  float*  XF = reinterpret_cast<float*>(smem_raw);   // [96][SF] x f32 -> tokens1
  float*  S  = XF + MR * SF;                         // [96][SF] t_sum / y / t_ffn
  float*  QH = S  + MR * SF;                         // q f32 -> t_sum bf16 -> h bf16
  __bf16* XB = reinterpret_cast<__bf16*>(QH + MR * SF); // [96][SB] x bf16 -> tokens1 bf16
  float*  TW = reinterpret_cast<float*>(XB + MR * SB);  // token-mlp weights (318 f32)

  const int tid  = threadIdx.x;
  const int lane = tid & 31;
  const int wave = tid >> 5;
  const int b0   = blockIdx.x * BT;

  // ---- phase 0: load x tile (coalesced) + token-mlp weights ----
  for (int e = tid; e < MR * CC; e += 256) {
    int r = e / CC, c = e - r * CC;
    int t = r >> 4, bi = r & 15;
    float v = x[((size_t)t * BSZ + b0 + bi) * CC + c];
    XF[r * SF + c] = v;
    XB[r * SB + c] = (__bf16)v;
  }
  for (int e = tid; e < 318; e += 256) {
    float v;
    if (e < 144)      v = tw1[e];
    else if (e < 168) v = tb1[e - 144];
    else if (e < 312) v = tw2[e - 168];
    else              v = tb2[e - 312];
    TW[e] = v;
  }
  __syncthreads();

  // ---- phase 1: token-mixing MLP (per (b,c): 6 -> 24 -> 6), into S ----
  // (reads XF, writes S)
  {
    const float* w1 = TW;       const float* b1 = TW + 144;
    const float* w2 = TW + 168; const float* b2 = TW + 312;
    for (int task = tid; task < BT * CC; task += 256) {
      int bi = task / CC, c = task - bi * CC;
      float u[TT];
#pragma unroll
      for (int t = 0; t < TT; ++t) u[t] = XF[(t * 16 + bi) * SF + c];
      float z[24];
#pragma unroll
      for (int j = 0; j < 24; ++j) {
        float s = b1[j];
#pragma unroll
        for (int t = 0; t < TT; ++t) s += u[t] * w1[t * 24 + j];
        z[j] = gelu_f(s);
      }
#pragma unroll
      for (int t = 0; t < TT; ++t) {
        float s = b2[t];
#pragma unroll
        for (int j = 0; j < 24; ++j) s += z[j] * w2[j * TT + t];
        S[(t * 16 + bi) * SF + c] = s;
      }
    }
  }
  // NOTE: no barrier here — phase 2 reads XB / writes QH, disjoint from
  // phase 1 (reads XF / writes S); lets token-MLP VALU overlap WMMA waves.

  // ---- phase 2: q = x @ q_w + q_b  (WMMA) -> QH (f32) ----
  run_gemm(XB, SB, CC / 32, qw_p, CC / 16, wave, lane,
           [&](int mt, int nt, v8f acc) {
             int n = lane & 15, h = lane >> 4;
             float bv = qb[nt * 16 + n];
#pragma unroll
             for (int g = 0; g < 8; ++g)
               QH[(mt * 16 + g + 8 * h) * SF + nt * 16 + n] = acc[g] + bv;
           });
  __syncthreads();

  // ---- phase 3: tiny attention per (b,h,t); keys==values==x; add into S ----
  {
    const float scale = 0.14433756729740643f;  // 48^-0.5
    for (int task = tid; task < BT * NH * TT; task += 256) {
      int t = task % TT;
      int h = (task / TT) % NH;
      int bi = task / (TT * NH);
      const float* qrow = &QH[(t * 16 + bi) * SF + h * DHD];
      float sc[TT]; float mx = -1e30f;
      for (int s = 0; s < TT; ++s) {
        const float* krow = &XF[(s * 16 + bi) * SF + h * DHD];
        float a = 0.f;
        for (int d = 0; d < DHD; ++d) a += qrow[d] * krow[d];
        a *= scale; sc[s] = a; mx = fmaxf(mx, a);
      }
      float sum = 0.f;
      for (int s = 0; s < TT; ++s) { sc[s] = expf(sc[s] - mx); sum += sc[s]; }
      float inv = 1.f / sum;
      for (int s = 0; s < TT; ++s) sc[s] *= inv;
      float* srow = &S[(t * 16 + bi) * SF + h * DHD];
      for (int d = 0; d < DHD; ++d) {
        float o = 0.f;
        for (int s = 0; s < TT; ++s) o += sc[s] * XF[(s * 16 + bi) * SF + h * DHD + d];
        srow[d] += o;
      }
    }
  }
  __syncthreads();

  // ---- phase 4: alpha gate (WMMA) ; t_sum*alp -> bf16 in QH region ----
  {
    __bf16* TS = reinterpret_cast<__bf16*>(QH);
    run_gemm(XB, SB, CC / 32, aw_p, CC / 16, wave, lane,
             [&](int mt, int nt, v8f acc) {
               int n = lane & 15, h = lane >> 4;
               float bv = ab[nt * 16 + n];
#pragma unroll
               for (int g = 0; g < 8; ++g) {
                 int row = mt * 16 + g + 8 * h, col = nt * 16 + n;
                 float a = acc[g] + bv;
                 float alp = fminf(fmaxf(a + 3.f, 0.f), 6.f) * (1.f / 3.f);
                 TS[row * SB + col] = (__bf16)(S[row * SF + col] * alp);
               }
             });
  }
  __syncthreads();

  // ---- phase 5: channel GEMM -> S ----
  run_gemm(reinterpret_cast<const __bf16*>(QH), SB, CC / 32, cw_p, CC / 16, wave, lane,
           [&](int mt, int nt, v8f acc) {
             int n = lane & 15, h = lane >> 4;
             float bv = cb[nt * 16 + n];
#pragma unroll
             for (int g = 0; g < 8; ++g)
               S[(mt * 16 + g + 8 * h) * SF + nt * 16 + n] = acc[g] + bv;
           });
  __syncthreads();

  // ---- phase 6: LN1: tokens1 = LN(x + S) -> XF (f32) and XB (bf16) ----
  if (tid < MR) {
    int r = tid;
    float sum = 0.f, sq = 0.f;
    for (int c = 0; c < CC; ++c) {
      float v = XF[r * SF + c] + S[r * SF + c];
      sum += v; sq += v * v;
    }
    float mu = sum * (1.f / CC);
    float var = sq * (1.f / CC) - mu * mu;
    float inv = rsqrtf(var + 1e-5f);
    for (int c = 0; c < CC; ++c) {
      float v = (XF[r * SF + c] + S[r * SF + c] - mu) * inv * lng[c] + lnb[c];
      XF[r * SF + c] = v;
      XB[r * SB + c] = (__bf16)v;
    }
  }
  __syncthreads();

  // ---- phase 7: FFN up: h = gelu(tokens1 @ ffn_w1 + b1) -> bf16 in QH region
  {
    __bf16* H = reinterpret_cast<__bf16*>(QH);
    run_gemm(XB, SB, CC / 32, f1_p, (2 * CC) / 16, wave, lane,
             [&](int mt, int nt, v8f acc) {
               int n = lane & 15, h = lane >> 4;
               float bv = f1b[nt * 16 + n];
#pragma unroll
               for (int g = 0; g < 8; ++g)
                 H[(mt * 16 + g + 8 * h) * SH + nt * 16 + n] =
                     (__bf16)gelu_f(acc[g] + bv);
             });
  }
  __syncthreads();

  // ---- phase 8: FFN down: t_ffn = h @ ffn_w2 + b2 -> S ----
  run_gemm(reinterpret_cast<const __bf16*>(QH), SH, (2 * CC) / 32, f2_p, CC / 16,
           wave, lane,
           [&](int mt, int nt, v8f acc) {
             int n = lane & 15, h = lane >> 4;
             float bv = f2b[nt * 16 + n];
#pragma unroll
             for (int g = 0; g < 8; ++g)
               S[(mt * 16 + g + 8 * h) * SF + nt * 16 + n] = acc[g] + bv;
           });
  __syncthreads();

  // ---- phase 9: LN2 in place into S, then coalesced store ----
  if (tid < MR) {
    int r = tid;
    float sum = 0.f, sq = 0.f;
    for (int c = 0; c < CC; ++c) {
      float v = XF[r * SF + c] + S[r * SF + c];
      sum += v; sq += v * v;
    }
    float mu = sum * (1.f / CC);
    float var = sq * (1.f / CC) - mu * mu;
    float inv = rsqrtf(var + 1e-5f);
    for (int c = 0; c < CC; ++c) {
      float v = (XF[r * SF + c] + S[r * SF + c] - mu) * inv * fng[c] + fnb[c];
      S[r * SF + c] = v;
    }
  }
  __syncthreads();
  for (int e = tid; e < MR * CC; e += 256) {
    int r = e / CC, c = e - r * CC;
    int t = r >> 4, bi = r & 15;
    out[((size_t)t * BSZ + b0 + bi) * CC + c] = S[r * SF + c];
  }
}

extern "C" void kernel_launch(void* const* d_in, const int* in_sizes, int n_in,
                              void* d_out, int out_size, void* d_ws, size_t ws_size,
                              hipStream_t stream) {
  (void)in_sizes; (void)n_in; (void)out_size; (void)ws_size;
  const float* x   = (const float*)d_in[0];
  const float* tw1 = (const float*)d_in[1];
  const float* tb1 = (const float*)d_in[2];
  const float* tw2 = (const float*)d_in[3];
  const float* tb2 = (const float*)d_in[4];
  const float* qw  = (const float*)d_in[5];
  const float* qb  = (const float*)d_in[6];
  const float* aw  = (const float*)d_in[7];
  const float* ab  = (const float*)d_in[8];
  const float* cw  = (const float*)d_in[9];
  const float* cb  = (const float*)d_in[10];
  const float* lng = (const float*)d_in[11];
  const float* lnb = (const float*)d_in[12];
  const float* f1w = (const float*)d_in[13];
  const float* f1b = (const float*)d_in[14];
  const float* f2w = (const float*)d_in[15];
  const float* f2b = (const float*)d_in[16];
  const float* fng = (const float*)d_in[17];
  const float* fnb = (const float*)d_in[18];

  __bf16* ws   = (__bf16*)d_ws;
  __bf16* qw_p = ws;                        // 192*192
  __bf16* aw_p = qw_p + 192 * 192;
  __bf16* cw_p = aw_p + 192 * 192;
  __bf16* f1_p = cw_p + 192 * 192;          // 192*384
  __bf16* f2_p = f1_p + 192 * 384;          // 384*192

  pack_w<<<(192 * 192 + 255) / 256, 256, 0, stream>>>(qw,  qw_p, 192, 192);
  pack_w<<<(192 * 192 + 255) / 256, 256, 0, stream>>>(aw,  aw_p, 192, 192);
  pack_w<<<(192 * 192 + 255) / 256, 256, 0, stream>>>(cw,  cw_p, 192, 192);
  pack_w<<<(192 * 384 + 255) / 256, 256, 0, stream>>>(f1w, f1_p, 192, 384);
  pack_w<<<(384 * 192 + 255) / 256, 256, 0, stream>>>(f2w, f2_p, 384, 192);

  constexpr size_t SMEM = (size_t)(3 * MR * SF) * 4 + (size_t)MR * SB * 2 + 320 * 4;
  mixer_block<<<BSZ / BT, 256, SMEM, stream>>>(
      x, tw1, tb1, tw2, tb2, qb, ab, cb, lng, lnb, f1b, f2b, fng, fnb,
      qw_p, aw_p, cw_p, f1_p, f2_p, (float*)d_out);
}
